// LatentAttentionBlock_21534966022741
// MI455X (gfx1250) — matrix operand order repaired and checked by reading, not compiled, not run on hardware
//
#include <hip/hip_runtime.h>

#define AS1 __attribute__((address_space(1)))
#define AS3 __attribute__((address_space(3)))

typedef __attribute__((ext_vector_type(16))) __bf16 v16bf;
typedef __attribute__((ext_vector_type(8)))  float  v8f;
typedef __attribute__((ext_vector_type(4)))  int    v4i;

union BF16x16 { v16bf v; v4i w[2]; };

__device__ __forceinline__ v8f wmma_bf16(v16bf a, v16bf b, v8f c) {
  return __builtin_amdgcn_wmma_f32_16x16x32_bf16(false, a, false, b, (short)0, c, false, false);
}

// LDS 128-bit load with explicit AS3 provenance -> guaranteed ds_load_b128
__device__ __forceinline__ v4i lds_b128(const void* p) {
  return *(const AS3 v4i*)p;
}

#if defined(__gfx1250__) && __has_builtin(__builtin_amdgcn_global_load_async_to_lds_b128)
#define HAVE_ASYNC_LDS 1
__device__ __forceinline__ void g2l_b128(const void* g, void* l) {
  // per-lane 16B async copy global -> LDS, tracked by ASYNCcnt
  AS1 v4i* gp = (AS1 v4i*)(unsigned long long)g;  // global pointers are numerically flat
  AS3 v4i* lp = (AS3 v4i*)l;                      // addrspacecast generic -> LDS
  __builtin_amdgcn_global_load_async_to_lds_b128(gp, lp, 0, 0);
}
__device__ __forceinline__ void wait_async0() {
#if __has_builtin(__builtin_amdgcn_s_wait_asynccnt)
  __builtin_amdgcn_s_wait_asynccnt(0);
#else
  asm volatile("s_wait_asynccnt 0x0" ::: "memory");
#endif
}
#else
#define HAVE_ASYNC_LDS 0
#endif

__device__ __forceinline__ float rmax16(float v) {
  v = fmaxf(v, __shfl_xor(v, 1, 32));
  v = fmaxf(v, __shfl_xor(v, 2, 32));
  v = fmaxf(v, __shfl_xor(v, 4, 32));
  v = fmaxf(v, __shfl_xor(v, 8, 32));
  return v;
}
__device__ __forceinline__ float rsum16(float v) {
  v += __shfl_xor(v, 1, 32);
  v += __shfl_xor(v, 2, 32);
  v += __shfl_xor(v, 4, 32);
  v += __shfl_xor(v, 8, 32);
  return v;
}

// ---------------- fp32 -> bf16 conversion (weights) ----------------
__global__ __launch_bounds__(256) void cvt_bf16(const float* __restrict__ in,
                                                __bf16* __restrict__ out, long n4) {
  long i = (long)blockIdx.x * 256 + threadIdx.x;
  if (i >= n4) return;
  float4 v = ((const float4*)in)[i];
  __bf16 o[4] = {(__bf16)v.x, (__bf16)v.y, (__bf16)v.z, (__bf16)v.w};
  *(uint2*)(out + i * 4) = *(const uint2*)o;
}

// ---------------- LayerNorm: one 256-thread block per row of 1024 ----------------
__global__ __launch_bounds__(256) void layernorm_k(const float* __restrict__ x,
                                                   const float* __restrict__ w,
                                                   __bf16* __restrict__ xn) {
  __shared__ float red[16];
  long m = blockIdx.x;
  const float4 v = ((const float4*)(x + m * 1024))[threadIdx.x];
  float s  = v.x + v.y + v.z + v.w;
  float s2 = v.x * v.x + v.y * v.y + v.z * v.z + v.w * v.w;
  for (int o = 1; o < 32; o <<= 1) { s += __shfl_xor(s, o, 32); s2 += __shfl_xor(s2, o, 32); }
  int wv = threadIdx.x >> 5, ln = threadIdx.x & 31;
  if (ln == 0) { red[wv] = s; red[8 + wv] = s2; }
  __syncthreads();
  if (threadIdx.x == 0) {
    float ts = 0.f, ts2 = 0.f;
    for (int i = 0; i < 8; i++) { ts += red[i]; ts2 += red[8 + i]; }
    red[0] = ts; red[8] = ts2;
  }
  __syncthreads();
  float mu  = red[0] * (1.0f / 1024.0f);
  float var = red[8] * (1.0f / 1024.0f) - mu * mu;
  float r   = rsqrtf(var + 1e-5f);
  int d = threadIdx.x * 4;
  __bf16 o[4] = {(__bf16)((v.x - mu) * r * w[d]),
                 (__bf16)((v.y - mu) * r * w[d + 1]),
                 (__bf16)((v.z - mu) * r * w[d + 2]),
                 (__bf16)((v.w - mu) * r * w[d + 3])};
  *(uint2*)(xn + m * 1024 + d) = *(const uint2*)o;
}

// ---------------- Tiled bf16 GEMM: C[M,N] = A[M,K] * Bw[N,K]^T ----------------
// MODE 0: bf16 store to outB.  MODE 1: outF = residual + acc.
template <int MODE>
__global__ __launch_bounds__(256) void gemm_bf16(const __bf16* __restrict__ A,
                                                 const __bf16* __restrict__ Bw,
                                                 __bf16* __restrict__ outB,
                                                 float* __restrict__ outF,
                                                 const float* __restrict__ residual,
                                                 int K, int ldOut) {
  __shared__ __bf16 Atile[2][128][32];
  __shared__ __bf16 Btile[2][128][32];
  const int t    = threadIdx.x;
  const int wave = t >> 5;
  const int lane = t & 31;
  const int half = lane >> 4;
  const int l16  = lane & 15;
  const long tileM = (long)blockIdx.x * 128;
  const long tileN = (long)blockIdx.y * 128;

  v8f acc[8] = {};

  const int row0 = t >> 2,         cg0 = t & 3;
  const int row1 = (t + 256) >> 2, cg1 = (t + 256) & 3;

  auto compute = [&](int buf) {
    BF16x16 afrag;
    const int arow = wave * 16 + l16;
    const int koff = half * 8;
    afrag.w[0] = lds_b128(&Atile[buf][arow][koff]);
    afrag.w[1] = lds_b128(&Atile[buf][arow][koff + 16]);
#pragma unroll
    for (int n = 0; n < 8; n++) {
      BF16x16 bfrag;
      bfrag.w[0] = lds_b128(&Btile[buf][n * 16 + l16][half * 8]);
      bfrag.w[1] = lds_b128(&Btile[buf][n * 16 + l16][half * 8 + 16]);
      acc[n] = wmma_bf16(afrag.v, bfrag.v, acc[n]);
    }
  };

#if HAVE_ASYNC_LDS
  auto stage = [&](int buf, int k0) {
    g2l_b128(A  + (tileM + row0) * K + k0 + cg0 * 8, &Atile[buf][row0][cg0 * 8]);
    g2l_b128(Bw + (tileN + row0) * K + k0 + cg0 * 8, &Btile[buf][row0][cg0 * 8]);
    g2l_b128(A  + (tileM + row1) * K + k0 + cg1 * 8, &Atile[buf][row1][cg1 * 8]);
    g2l_b128(Bw + (tileN + row1) * K + k0 + cg1 * 8, &Btile[buf][row1][cg1 * 8]);
  };
  stage(0, 0);
  int it = 0;
  for (int k0 = 0; k0 < K; k0 += 32, ++it) {
    const int buf = it & 1;
    wait_async0();
    __syncthreads();
    if (k0 + 32 < K) stage(buf ^ 1, k0 + 32);
    compute(buf);
  }
#else
  uint4 ta[2], tb[2];
  auto loadtemps = [&](int k0) {
    ta[0] = *(const uint4*)(A  + (tileM + row0) * K + k0 + cg0 * 8);
    tb[0] = *(const uint4*)(Bw + (tileN + row0) * K + k0 + cg0 * 8);
    ta[1] = *(const uint4*)(A  + (tileM + row1) * K + k0 + cg1 * 8);
    tb[1] = *(const uint4*)(Bw + (tileN + row1) * K + k0 + cg1 * 8);
  };
  loadtemps(0);
  int it = 0;
  for (int k0 = 0; k0 < K; k0 += 32, ++it) {
    const int buf = it & 1;
    *(uint4*)&Atile[buf][row0][cg0 * 8] = ta[0];
    *(uint4*)&Btile[buf][row0][cg0 * 8] = tb[0];
    *(uint4*)&Atile[buf][row1][cg1 * 8] = ta[1];
    *(uint4*)&Btile[buf][row1][cg1 * 8] = tb[1];
    if (k0 + 32 < K) loadtemps(k0 + 32);
    __syncthreads();
    compute(buf);
  }
#endif

#pragma unroll
  for (int n = 0; n < 8; n++) {
    long col = tileN + n * 16 + l16;
#pragma unroll
    for (int j = 0; j < 8; j++) {
      long row = tileM + wave * 16 + j + half * 8;
      float v = acc[n][j];
      if (MODE == 0) {
        outB[row * (long)ldOut + col] = (__bf16)v;
      } else {
        long o = row * (long)ldOut + col;
        outF[o] = residual[o] + v;
      }
    }
  }
}

// ---------------- GeGLU: lin * gelu(pre); split into cat-local and value ----------------
__global__ __launch_bounds__(256) void geglu_k(const __bf16* __restrict__ proj,
                                               __bf16* __restrict__ catbuf,
                                               __bf16* __restrict__ vbuf) {
  long m = blockIdx.x;
  for (int e = threadIdx.x; e < 2048; e += 256) {
    float lin = (float)proj[m * 4352 + 256 + e];
    float pre = (float)proj[m * 4352 + 2304 + e];
    float g   = lin * (0.5f * pre * (1.0f + erff(pre * 0.70710678118654752f)));
    if (e < 1024) catbuf[m * 2048 + e] = (__bf16)g;
    else          vbuf[m * 1024 + (e - 1024)] = (__bf16)g;
  }
}

// ---------------- Flash attention: 4 waves/block, 16 q-rows each ----------------
__global__ __launch_bounds__(128) void attn_k(const __bf16* __restrict__ proj,
                                              const __bf16* __restrict__ vbuf,
                                              __bf16* __restrict__ catbuf,
                                              const float* __restrict__ pbm_p,
                                              const int* __restrict__ fi_p,
                                              const int* __restrict__ li_p) {
  __shared__ __bf16 ktile[32][16];     // [key][dh]
  __shared__ __bf16 vtileT[128][32];   // [dh][key]  (transposed V)
  __shared__ __bf16 ptile[4][16][32];  // per-wave probs [qrow][key]
  const int t    = threadIdx.x;
  const int wave = t >> 5;
  const int lane = t & 31;
  const int half = lane >> 4;
  const int l16  = lane & 15;
  const int b = blockIdx.y >> 3;
  const int h = blockIdx.y & 7;
  const int S = 1024;
  const long mbase = (long)b * S;

  float spx = *pbm_p;
  float sp  = (spx > 20.f) ? spx : log1pf(__expf(spx));
  int fi = *fi_p; if (fi >= S) fi = 0;
  int li = *li_p; if (li >= S) li = 0;

  // Q fragment (dh_q=16 zero-padded to K=32)
  const int q_s = blockIdx.x * 64 + wave * 16 + l16;
  BF16x16 qfrag;
#pragma unroll
  for (int i = 0; i < 16; i++) {
    int Kd = i + half * 16;
    qfrag.v[i] = (Kd < 16) ? proj[(mbase + q_s) * 4352 + h * 16 + Kd] : (__bf16)0.0f;
  }

  v8f oacc[8] = {};
  float mrow[8], lrow[8];
#pragma unroll
  for (int j = 0; j < 8; j++) { mrow[j] = -3.0e38f; lrow[j] = 0.0f; }

#pragma unroll 1
  for (int kb = 0; kb < 32; kb++) {
    const int kbase = kb * 32;
    // stage K tile (32x16 bf16): 32B contiguous per row
#if HAVE_ASYNC_LDS
    if (t < 64) {
      int key = t >> 1, hh = (t & 1) * 8;
      g2l_b128(&proj[(mbase + kbase + key) * 4352 + 128 + h * 16 + hh], &ktile[key][hh]);
    }
#else
    if (t < 64) {
      int key = t >> 1, hh = (t & 1) * 8;
      *(uint4*)&ktile[key][hh] =
          *(const uint4*)&proj[(mbase + kbase + key) * 4352 + 128 + h * 16 + hh];
    }
#endif
    // stage V transposed: b128 global loads, b16 LDS scatter
#pragma unroll
    for (int i = 0; i < 4; i++) {
      int idx = t + i * 128;           // 0..511
      int key = idx >> 4;              // 0..31
      int dh8 = (idx & 15) * 8;        // 0..120
      uint4 vv = *(const uint4*)&vbuf[(mbase + kbase + key) * 1024 + h * 128 + dh8];
      const __bf16* pv = (const __bf16*)&vv;
#pragma unroll
      for (int q = 0; q < 8; q++) vtileT[dh8 + q][key] = pv[q];
    }
#if HAVE_ASYNC_LDS
    wait_async0();
#endif
    __syncthreads();

    // scores = Q * K^T for keys kbase..kbase+31 (upper-half K rows are zero pad)
    BF16x16 kf0, kf1;
    {
      v4i z = {0, 0, 0, 0};
      v4i a0 = lds_b128(&ktile[l16][0]);
      v4i a1 = lds_b128(&ktile[l16][8]);
      v4i b0 = lds_b128(&ktile[l16 + 16][0]);
      v4i b1 = lds_b128(&ktile[l16 + 16][8]);
      kf0.w[0] = half ? z : a0; kf0.w[1] = half ? z : a1;
      kf1.w[0] = half ? z : b0; kf1.w[1] = half ? z : b1;
    }
    v8f sc0 = {}, sc1 = {};
    sc0 = wmma_bf16(qfrag.v, kf0.v, sc0);
    sc1 = wmma_bf16(qfrag.v, kf1.v, sc1);

    // mask + online softmax (rows: j + half*8, cols: l16 / l16+16)
#pragma unroll
    for (int j = 0; j < 8; j++) {
      int qrow = blockIdx.x * 64 + wave * 16 + j + half * 8;
      int k0s = kbase + l16, k1s = kbase + 16 + l16;
      float v0 = sc0[j] * 0.25f, v1 = sc1[j] * 0.25f;
      bool vis0 = (k0s <= qrow) || ((qrow >= fi) && (k0s >= li));
      bool vis1 = (k1s <= qrow) || ((qrow >= fi) && (k1s >= li));
      v0 = vis0 ? v0 + sp * (float)(k0s - qrow) : -1e30f;
      v1 = vis1 ? v1 + sp * (float)(k1s - qrow) : -1e30f;
      float mn = fmaxf(mrow[j], rmax16(fmaxf(v0, v1)));
      float p0 = __expf(v0 - mn), p1 = __expf(v1 - mn);
      float f  = __expf(mrow[j] - mn);
      lrow[j] = lrow[j] * f + rsum16(p0 + p1);
      mrow[j] = mn;
#pragma unroll
      for (int n = 0; n < 8; n++) oacc[n][j] *= f;
      ptile[wave][j + half * 8][l16]      = (__bf16)p0;
      ptile[wave][j + half * 8][l16 + 16] = (__bf16)p1;
    }

    // O += P * V
    BF16x16 pfrag;
    pfrag.w[0] = lds_b128(&ptile[wave][l16][half * 8]);
    pfrag.w[1] = lds_b128(&ptile[wave][l16][half * 8 + 16]);
#pragma unroll
    for (int n = 0; n < 8; n++) {
      BF16x16 vfrag;
      vfrag.w[0] = lds_b128(&vtileT[n * 16 + l16][half * 8]);
      vfrag.w[1] = lds_b128(&vtileT[n * 16 + l16][half * 8 + 16]);
      oacc[n] = wmma_bf16(pfrag.v, vfrag.v, oacc[n]);
    }
    __syncthreads();
  }

#pragma unroll
  for (int n = 0; n < 8; n++) {
#pragma unroll
    for (int j = 0; j < 8; j++) {
      int qrow = blockIdx.x * 64 + wave * 16 + j + half * 8;
      float v = oacc[n][j] / lrow[j];
      catbuf[(mbase + qrow) * 2048 + 1024 + h * 128 + n * 16 + l16] = (__bf16)v;
    }
  }
}

extern "C" void kernel_launch(void* const* d_in, const int* in_sizes, int n_in,
                              void* d_out, int out_size, void* d_ws, size_t ws_size,
                              hipStream_t stream) {
  (void)in_sizes; (void)n_in; (void)out_size; (void)ws_size;
  const float* x         = (const float*)d_in[0];
  const float* norm_w    = (const float*)d_in[1];
  const float* expand_w  = (const float*)d_in[2];   // (4352, 1024)
  const float* project_w = (const float*)d_in[3];   // (1024, 2048)
  const float* pbm       = (const float*)d_in[4];
  const int*   fi        = (const int*)d_in[5];
  const int*   li        = (const int*)d_in[6];
  float* out = (float*)d_out;

  const long M = 16384;  // B*S
  char* ws = (char*)d_ws;
  __bf16* xn      = (__bf16*)ws;              ws += M * 1024 * 2;
  __bf16* wexp    = (__bf16*)ws;              ws += 4352L * 1024 * 2;
  __bf16* wproj   = (__bf16*)ws;              ws += 1024L * 2048 * 2;
  __bf16* projbuf = (__bf16*)ws;              ws += M * 4352 * 2;
  __bf16* catbuf  = (__bf16*)ws;              ws += M * 2048 * 2;
  __bf16* vbuf    = (__bf16*)ws;              ws += M * 1024 * 2;

  // weights -> bf16
  cvt_bf16<<<(4352L * 1024 / 4 + 255) / 256, 256, 0, stream>>>(expand_w, wexp, 4352L * 1024 / 4);
  cvt_bf16<<<(1024L * 2048 / 4 + 255) / 256, 256, 0, stream>>>(project_w, wproj, 1024L * 2048 / 4);

  // layernorm
  layernorm_k<<<(int)M, 256, 0, stream>>>(x, norm_w, xn);

  // expand: proj[M,4352] = xn[M,1024] @ wexp[4352,1024]^T
  gemm_bf16<0><<<dim3(128, 34), 256, 0, stream>>>(xn, wexp, projbuf, nullptr, nullptr,
                                                  1024, 4352);

  // geglu -> cat[:, :1024] and value
  geglu_k<<<(int)M, 256, 0, stream>>>(projbuf, catbuf, vbuf);

  // attention -> cat[:, 1024:]
  attn_k<<<dim3(16, 128), 128, 0, stream>>>(projbuf, vbuf, catbuf, pbm, fi, li);

  // project + residual: out[M,1024] = x + cat[M,2048] @ wproj[1024,2048]^T
  gemm_bf16<1><<<dim3(128, 8), 256, 0, stream>>>(catbuf, wproj, nullptr, out, x,
                                                 2048, 1024);
}
